// MipMap_63591285785200
// MI455X (gfx1250) — compile-verified, matrix operand order checked by reading, not run on hardware
//
#include <hip/hip_runtime.h>

// ---------------------------------------------------------------------------
// Types for CDNA5 WMMA / TDM
// ---------------------------------------------------------------------------
typedef __attribute__((ext_vector_type(16))) _Float16 v16h;
typedef __attribute__((ext_vector_type(8)))  _Float16 half8;
typedef __attribute__((ext_vector_type(8)))  float    v8f;
typedef __attribute__((ext_vector_type(4)))  unsigned int u32x4;
typedef __attribute__((ext_vector_type(4)))  int      i32x4v;
typedef __attribute__((ext_vector_type(8)))  int      i32x8v;

union V16U { v16h v; half8 h[2]; };

#if defined(__has_builtin)
#if __has_builtin(__builtin_amdgcn_tensor_load_to_lds) && __has_builtin(__builtin_amdgcn_s_wait_tensorcnt)
#define HAVE_TDM 1
#endif
#endif

// ---------------------------------------------------------------------------
// Problem constants
// ---------------------------------------------------------------------------
#define RES   128
#define FEAT  64
#define NPTS  131072
// strides in floats for (x, y, z, f) C-order grid
#define SX    (128 * 128 * 64)   // 1048576
#define SY    (128 * 64)         // 8192
#define SZ    64

// Gaussian(std=1, M=5), normalized (computed offline in double):
__device__ __constant__ float GW[5] = {
    0.05448868454f, 0.24420134159f, 0.40261994772f, 0.24420134159f, 0.05448868454f
};

__device__ __forceinline__ int reflect128(int j) {
    j = (j < 0) ? -j : j;
    return (j > 127) ? (254 - j) : j;
}

// ---------------------------------------------------------------------------
// Kernel 1: blur along x (stride SX). One thread = one float4 feature group.
// Pure streaming, memory-bound (~1.07 GB traffic).
// ---------------------------------------------------------------------------
__global__ void blur_x(const float* __restrict__ in, float* __restrict__ out) {
    long tid = (long)blockIdx.x * blockDim.x + threadIdx.x;   // 128^3 * 16
    int  fg  = (int)(tid & 15);
    long vox = tid >> 4;
    int  z   = (int)(vox & 127);
    int  y   = (int)((vox >> 7) & 127);
    int  x   = (int)(vox >> 14);

    long off_base = (long)x * SX + (long)y * SY + (long)z * SZ;
    long off_no   = off_base - (long)x * SX + fg * 4;

    float r0 = 0.f, r1 = 0.f, r2 = 0.f, r3 = 0.f;
#pragma unroll
    for (int t = 0; t < 5; ++t) {
        int j = reflect128(x + t - 2);
        const float4 v = *(const float4*)(in + off_no + (long)j * SX);
        float w = GW[t];
        r0 += w * v.x; r1 += w * v.y; r2 += w * v.z; r3 += w * v.w;
    }
    float4 res; res.x = r0; res.y = r1; res.z = r2; res.w = r3;
    *(float4*)(out + off_base + fg * 4) = res;
}

// ---------------------------------------------------------------------------
// Kernel 2: FUSED y+z blur through LDS. Block = 256 threads.
// Tile: one x-slice, 16x16 (y,z) output tile, 16-feature group.
//  - stage 20x20x16 input halo tile in LDS (25.6 KB)
//  - y-blur -> 16x20x16 intermediate (20.5 KB)
//  - z-blur -> 16x16x16 output
// Saves one full 1.07 GB grid read+write vs separate passes.
// ---------------------------------------------------------------------------
__global__ void blur_yz_fused(const float* __restrict__ in, float* __restrict__ out) {
    __shared__ float in_t [20 * 20 * 16];
    __shared__ float mid_t[16 * 20 * 16];

    const int b  = blockIdx.x;
    const int fg = b & 3;            // feature group of 16
    const int zt = (b >> 2) & 7;
    const int yt = (b >> 5) & 7;
    const int x  = b >> 8;           // 0..127
    const int y0 = yt * 16;
    const int z0 = zt * 16;
    const int tid = threadIdx.x;

    const long base_x = (long)x * SX + fg * 16;

    // stage input tile with reflect halo
    for (int e = tid; e < 20 * 20 * 16; e += 256) {
        int f  = e & 15;
        int v  = e >> 4;             // 0..399
        int zz = v % 20;
        int yy = v / 20;
        int gy = reflect128(y0 + yy - 2);
        int gz = reflect128(z0 + zz - 2);
        in_t[e] = in[base_x + (long)gy * SY + (long)gz * SZ + f];
    }
    __syncthreads();

    // y-blur
    for (int e = tid; e < 16 * 20 * 16; e += 256) {
        int f  = e & 15;
        int v  = e >> 4;             // 0..319
        int zz = v % 20;
        int yy = v / 20;
        float s = 0.f;
#pragma unroll
        for (int t = 0; t < 5; ++t)
            s += GW[t] * in_t[((yy + t) * 20 + zz) * 16 + f];
        mid_t[e] = s;
    }
    __syncthreads();

    // z-blur + store
    for (int e = tid; e < 16 * 16 * 16; e += 256) {
        int f  = e & 15;
        int zz = (e >> 4) & 15;
        int yy = e >> 8;
        float s = 0.f;
#pragma unroll
        for (int t = 0; t < 5; ++t)
            s += GW[t] * mid_t[(yy * 20 + zz + t) * 16 + f];
        out[base_x + (long)(y0 + yy) * SY + (long)(z0 + zz) * SZ + f] = s;
    }
}

// ---------------------------------------------------------------------------
// Kernel 3: pre-swizzle W1 (64x128) and W2 (128x4 -> 128x16 pad) into the
// f16 WMMA B-operand lane/half layout:
//   element (k_local, n): lane = n + 16*((k_local>>3)&1),
//                         half = (k_local&7) + 8*(k_local>>4)
// Stored [chunk][lane][half] -> each lane loads a contiguous 32B v16h.
// ---------------------------------------------------------------------------
__global__ void prep_weights(const float* __restrict__ W1, const float* __restrict__ W2,
                             _Float16* __restrict__ W1h, _Float16* __restrict__ W2h) {
    for (int idx = threadIdx.x; idx < 8192; idx += blockDim.x) {
        int half = idx & 15;
        int lane = (idx >> 4) & 31;
        int cbkc = idx >> 9;           // 0..15
        int cb   = cbkc >> 1;
        int kc   = cbkc & 1;
        int n    = lane & 15;
        int kl   = (half & 7) + 8 * ((lane >> 4) & 1) + 16 * (half >> 3);
        int k    = kc * 32 + kl;       // 0..63
        W1h[idx] = (_Float16)W1[k * 128 + cb * 16 + n];
    }
    for (int idx = threadIdx.x; idx < 2048; idx += blockDim.x) {
        int half = idx & 15;
        int lane = (idx >> 4) & 31;
        int kc   = idx >> 9;           // 0..3
        int n    = lane & 15;
        int kl   = (half & 7) + 8 * ((lane >> 4) & 1) + 16 * (half >> 3);
        int k    = kc * 32 + kl;       // 0..127
        W2h[idx] = (n < 4) ? (_Float16)W2[k * 4 + n] : (_Float16)0.0f;
    }
}

// ---------------------------------------------------------------------------
// Kernel 4: trilinear interp + MLP. Block = 128 threads (4 waves), 64 points.
// Wave owns 16 points; cooperative gather (coalesced 256B/corner); WMMA
// f32_16x16x32_f16 both layers. Weights DMA'd to LDS by the Tensor Data
// Mover (overlapped with the gather phase).
// ---------------------------------------------------------------------------
__device__ __forceinline__ float lerpf(float a, float b, float t) {
    return (1.0f - t) * a + t * b;
}

__global__ void interp_mlp(const float* __restrict__ pt,
                           const float* __restrict__ fmap,
                           const _Float16* __restrict__ W1h,   // 8192 halves
                           const float* __restrict__ b1,
                           const _Float16* __restrict__ W2h,   // 2048 halves (== W1h+8192)
                           const float* __restrict__ b2,
                           float* __restrict__ out) {
    __shared__ _Float16 featT[64 * 64];    // [point-in-block][feature]
    __shared__ _Float16 hT[64 * 128];      // [point-in-block][hidden]
#ifdef HAVE_TDM
    __shared__ _Float16 Wlds[10240];       // W1h(8192) ++ W2h(2048)
#endif

    const int lane    = threadIdx.x & 31;
    const int wv      = threadIdx.x >> 5;
    const int rowBase = wv * 16;
    const int pBase   = blockIdx.x * 64 + rowBase;
    const int f0      = lane * 2;

#ifdef HAVE_TDM
    // ---- TDM: DMA the 20 KB pre-swizzled weights into LDS (wave 0 issues;
    // overlaps with the gather phase below). D# per CDNA5 ISA.
    if (wv == 0) {
        unsigned ldsOff = (unsigned)(unsigned long long)(&Wlds[0]); // flat[31:0] == LDS offset
        unsigned long long ga = (unsigned long long)W1h;
        const unsigned TOT = 10240;                    // elements (data_size = 2B)
        u32x4 g0 = { 1u,                               // count=1, user mode
                     ldsOff,
                     (unsigned)(ga & 0xFFFFFFFFull),
                     (unsigned)(((ga >> 32) & 0x1FFFFFFull) | (2u << 30)) }; // type=2
        i32x8v g1 = { (int)(1u << 16),                 // data_size=1 (2B), mask=0
                      (int)((TOT & 0xFFFFu) << 16),    // tensor_dim0 lo
                      (int)((TOT >> 16) | (1u << 16)), // tensor_dim0 hi | tensor_dim1=1
                      (int)((TOT & 0xFFFFu) << 16),    // tile_dim0
                      0,                               // tile_dim1/2 = 0
                      (int)TOT,                        // tensor_dim0_stride lo
                      0, 0 };
        i32x4v g2 = { 0, 0, 0, 0 };
        i32x4v g3 = { 0, 0, 0, 0 };
        i32x8v g4 = { 0, 0, 0, 0, 0, 0, 0, 0 };       // extra group (clang-23 6-arg form)
        __builtin_amdgcn_tensor_load_to_lds(g0, g1, g2, g3, g4, 0);
    }
#endif

    // ---- load the wave's 16 point coords once (lanes 0..15), broadcast later
    float cx = 0.f, cy = 0.f, cz = 0.f;
    if (lane < 16) {
        int p = pBase + lane;
        cx = pt[3 * p + 0];
        cy = pt[3 * p + 1];
        cz = pt[3 * p + 2];
    }

    // ---- trilinear interpolation: one point per iteration, wave-cooperative
    for (int i = 0; i < 16; ++i) {
        float px = __shfl(cx, i);
        float py = __shfl(cy, i);
        float pz = __shfl(cz, i);

        float axf = (px + 1.0f) * 0.5f * 127.0f;
        float ayf = (py + 1.0f) * 0.5f * 127.0f;
        float azf = (pz + 1.0f) * 0.5f * 127.0f;
        int ix = (int)axf, iy = (int)ayf, iz = (int)azf;
        float fx = axf - ix, fy = ayf - iy, fz = azf - iz;
        int ix1 = ix + 1 < 127 ? ix + 1 : 127;
        int iy1 = iy + 1 < 127 ? iy + 1 : 127;
        int iz1 = iz + 1 < 127 ? iz + 1 : 127;

        long bx0 = (long)ix  * SX, bx1 = (long)ix1 * SX;
        long by0 = (long)iy  * SY, by1 = (long)iy1 * SY;
        long bz0 = (long)iz  * SZ, bz1 = (long)iz1 * SZ;
        const float* g = fmap + f0;

        // prefetch next point's extreme corners (global_prefetch_b8)
        if (i < 15) {
            float nx = __shfl(cx, i + 1), ny = __shfl(cy, i + 1), nz = __shfl(cz, i + 1);
            int jx = (int)((nx + 1.0f) * 0.5f * 127.0f);
            int jy = (int)((ny + 1.0f) * 0.5f * 127.0f);
            int jz = (int)((nz + 1.0f) * 0.5f * 127.0f);
            const float* pf = g + (long)jx * SX + (long)jy * SY + (long)jz * SZ;
            __builtin_prefetch(pf, 0, 1);
            __builtin_prefetch(pf + SX + SY + SZ, 0, 1);
        }

        float2 c000 = *(const float2*)(g + bx0 + by0 + bz0);
        float2 c001 = *(const float2*)(g + bx0 + by0 + bz1);
        float2 c010 = *(const float2*)(g + bx0 + by1 + bz0);
        float2 c011 = *(const float2*)(g + bx0 + by1 + bz1);
        float2 c100 = *(const float2*)(g + bx1 + by0 + bz0);
        float2 c101 = *(const float2*)(g + bx1 + by0 + bz1);
        float2 c110 = *(const float2*)(g + bx1 + by1 + bz0);
        float2 c111 = *(const float2*)(g + bx1 + by1 + bz1);

        // lerp x, then y, then z (matches reference order)
        float a00x = lerpf(c000.x, c100.x, fx), a00y = lerpf(c000.y, c100.y, fx);
        float a01x = lerpf(c001.x, c101.x, fx), a01y = lerpf(c001.y, c101.y, fx);
        float a10x = lerpf(c010.x, c110.x, fx), a10y = lerpf(c010.y, c110.y, fx);
        float a11x = lerpf(c011.x, c111.x, fx), a11y = lerpf(c011.y, c111.y, fx);
        float b0x  = lerpf(a00x, a10x, fy),     b0y  = lerpf(a00y, a10y, fy);
        float b1x_ = lerpf(a01x, a11x, fy),     b1y_ = lerpf(a01y, a11y, fy);
        float rx   = lerpf(b0x, b1x_, fz),      ry   = lerpf(b0y, b1y_, fz);

        int base = (rowBase + i) * 64 + f0;
        featT[base + 0] = (_Float16)rx;
        featT[base + 1] = (_Float16)ry;
    }
    __syncthreads();

#ifdef HAVE_TDM
    if (wv == 0) __builtin_amdgcn_s_wait_tensorcnt(0);
    __syncthreads();                       // weights visible to all waves
    const _Float16* Wb1 = &Wlds[0];
    const _Float16* Wb2 = &Wlds[8192];
#else
    const _Float16* Wb1 = W1h;
    const _Float16* Wb2 = W2h;
#endif

    // ---- A operands (ISA 16-bit A 16x32 layout):
    // lane<16: row=lane,    K = {hb..hb+7, hb+16..hb+23}, hb=0
    // lane>=16: row=lane-16, hb=8
    const int rA = rowBase + (lane & 15);
    const int hb = (lane >> 4) * 8;

    V16U a0, a1;
    a0.h[0] = *(const half8*)&featT[rA * 64 +  0 + hb];
    a0.h[1] = *(const half8*)&featT[rA * 64 + 16 + hb];
    a1.h[0] = *(const half8*)&featT[rA * 64 + 32 + hb];
    a1.h[1] = *(const half8*)&featT[rA * 64 + 48 + hb];

    const int ncol = lane & 15;

    // ---- layer 1: feat(16x64) @ W1(64x128), 8 col blocks x 2 K chunks
    for (int cb = 0; cb < 8; ++cb) {
        V16U bm0, bm1;
        bm0.h[0] = *(const half8*)(Wb1 + ((cb * 2 + 0) * 32 + lane) * 16);
        bm0.h[1] = *(const half8*)(Wb1 + ((cb * 2 + 0) * 32 + lane) * 16 + 8);
        bm1.h[0] = *(const half8*)(Wb1 + ((cb * 2 + 1) * 32 + lane) * 16);
        bm1.h[1] = *(const half8*)(Wb1 + ((cb * 2 + 1) * 32 + lane) * 16 + 8);
        v8f acc = {0.f, 0.f, 0.f, 0.f, 0.f, 0.f, 0.f, 0.f};
        acc = __builtin_amdgcn_wmma_f32_16x16x32_f16(false, a0.v, false, bm0.v,
                                                     (short)0, acc, false, false);
        acc = __builtin_amdgcn_wmma_f32_16x16x32_f16(false, a1.v, false, bm1.v,
                                                     (short)0, acc, false, false);
        float bb   = b1[cb * 16 + ncol];
        int   colg = cb * 16 + ncol;
#pragma unroll
        for (int r = 0; r < 8; ++r) {
            // D layout: VGPR r holds row m = r + 8*(lane>=16), col = lane&15
            float v = acc[r] + bb;
            v = v > 0.0f ? v : 0.0f;
            hT[(rowBase + r + hb) * 128 + colg] = (_Float16)v;
        }
    }
    __syncthreads();

    // ---- layer 2: h(16x128) @ W2pad(128x16), cols 0..3 valid
    v8f acc2 = {0.f, 0.f, 0.f, 0.f, 0.f, 0.f, 0.f, 0.f};
    for (int kc = 0; kc < 4; ++kc) {
        V16U a2, bm;
        a2.h[0] = *(const half8*)&hT[rA * 128 + kc * 32 +  0 + hb];
        a2.h[1] = *(const half8*)&hT[rA * 128 + kc * 32 + 16 + hb];
        bm.h[0] = *(const half8*)(Wb2 + (kc * 32 + lane) * 16);
        bm.h[1] = *(const half8*)(Wb2 + (kc * 32 + lane) * 16 + 8);
        acc2 = __builtin_amdgcn_wmma_f32_16x16x32_f16(false, a2.v, false, bm.v,
                                                      (short)0, acc2, false, false);
    }

    if (ncol < 4) {
        float bb2 = b2[ncol];
#pragma unroll
        for (int r = 0; r < 8; ++r) {
            int p = pBase + r + hb;
            out[p * 4 + ncol] = acc2[r] + bb2;
        }
    }
}

// ---------------------------------------------------------------------------
// Host-side launcher
// ---------------------------------------------------------------------------
extern "C" void kernel_launch(void* const* d_in, const int* in_sizes, int n_in,
                              void* d_out, int out_size, void* d_ws, size_t ws_size,
                              hipStream_t stream) {
    const float* pt   = (const float*)d_in[0];   // N x 3
    const float* base = (const float*)d_in[1];   // 128^3 x 64
    const float* W1   = (const float*)d_in[2];   // 64 x 128
    const float* b1   = (const float*)d_in[3];   // 128
    const float* W2   = (const float*)d_in[4];   // 128 x 4
    const float* b2   = (const float*)d_in[5];   // 4
    float*       out  = (float*)d_out;           // N x 4

    const size_t GRID_ELEMS = (size_t)RES * RES * RES * FEAT;  // 134,217,728 floats
    float* ws0 = (float*)d_ws;
    float* ws1 = ws0 + GRID_ELEMS;
    _Float16* W1h = (_Float16*)((char*)d_ws + 2 * GRID_ELEMS * sizeof(float));
    _Float16* W2h = W1h + 8192;

    // Blur: x pass (streaming), then fused y+z pass through LDS. fmap -> ws1.
    const long total = (long)RES * RES * RES * (FEAT / 4);     // 33,554,432 threads
    blur_x<<<dim3((unsigned)(total / 256)), dim3(256), 0, stream>>>(base, ws0);
    blur_yz_fused<<<dim3(32768), dim3(256), 0, stream>>>(ws0, ws1);

    // Weight pre-swizzle into WMMA B-operand layout (f16)
    prep_weights<<<1, 256, 0, stream>>>(W1, W2, W1h, W2h);

    // Interp + MLP: 64 points per 128-thread block
    interp_mlp<<<NPTS / 64, 128, 0, stream>>>(pt, ws1, W1h, b1, W2h, b2, out);
}
